// KVAdapterInjector_20615843021107
// MI455X (gfx1250) — matrix operand order repaired and checked by reading, not compiled, not run on hardware
//
#include <hip/hip_runtime.h>

// ---------------- problem constants ----------------
#define T_SEQ   2048
#define D_MODEL 4096
#define NH      32
#define NKV     8
#define HD      128
#define RV      64
#define RANK    16
#define KV_LEN  (RV + T_SEQ)   // 2112
#define GQA     (NH / NKV)     // 4

// ---------------- WMMA / vector types ----------------
typedef __bf16        v16bf __attribute__((ext_vector_type(16)));
typedef float         v8f   __attribute__((ext_vector_type(8)));
typedef unsigned int  v4u   __attribute__((ext_vector_type(4)));

union FragBF { v16bf v; unsigned short u[16]; v4u q[2]; };
union Pack8  { v4u q; unsigned short s[8]; };

__device__ __forceinline__ unsigned short f2bf(float f) {
    unsigned int u = __float_as_uint(f);
    u += 0x7FFFu + ((u >> 16) & 1u);   // round-to-nearest-even
    return (unsigned short)(u >> 16);
}

// CDNA5 async global->LDS copy: 16B per lane, tracked by ASYNCcnt.
// ldsoff = LDS byte offset (low 32 bits of generic pointer to __shared__).
__device__ __forceinline__ void async_copy_b128(unsigned int ldsoff, const void* gaddr) {
    asm volatile("global_load_async_to_lds_b128 %0, %1, off"
                 :: "v"(ldsoff), "v"(gaddr) : "memory");
}
__device__ __forceinline__ void async_wait0() {
    asm volatile("s_wait_asynccnt 0x0" ::: "memory");
}

// ---------------- 0. f32 -> bf16 convert ----------------
__global__ void k_cvt_bf16(const float* __restrict__ src,
                           unsigned short* __restrict__ dst, int n) {
    int i = blockIdx.x * blockDim.x + threadIdx.x;
    int stride = gridDim.x * blockDim.x;
    for (; i < n; i += stride) dst[i] = f2bf(src[i]);
}

// ---------------- 1. LoRA adapter on virtual KV ----------------
__global__ void k_lora_adapt(const float* __restrict__ kv,     // (NKV,RV,HD)
                             const float* __restrict__ Am,     // (HD,RANK)
                             const float* __restrict__ Bm,     // (RANK,HD)
                             const float* __restrict__ scale,  // scalar
                             unsigned short* __restrict__ dst) // (NKV,KV_LEN,HD)
{
    __shared__ float xs[HD];
    __shared__ float ts[RANK];
    int blk = blockIdx.x;             // NKV*RV blocks
    int h = blk / RV, r = blk % RV;
    int i = threadIdx.x;              // 128 threads
    xs[i] = kv[(h * RV + r) * HD + i];
    __syncthreads();
    if (i < RANK) {
        float acc = 0.f;
        for (int k = 0; k < HD; ++k) acc += xs[k] * Am[k * RANK + i];
        ts[i] = acc;
    }
    __syncthreads();
    float s = *scale;
    float acc = 0.f;
    #pragma unroll
    for (int j = 0; j < RANK; ++j) acc += ts[j] * Bm[j * HD + i];
    dst[(h * KV_LEN + r) * HD + i] = f2bf(xs[i] + s * acc);
}

// ---------------- 2. bf16 WMMA GEMM: C_f32[M,N] = A[M,K] * B[K,N] ----------------
#define BM  128
#define BN  128
#define BK  64
#define LDA 72     // shorts; 144 B rows (16B aligned)
#define LDBT 72

__global__ __launch_bounds__(256) void k_gemm_bf16(
    const unsigned short* __restrict__ A,
    const unsigned short* __restrict__ B,
    float* __restrict__ C, int M, int N, int K)
{
    __shared__ unsigned short As[BM * LDA];     // 18432 B
    __shared__ unsigned short BsT[BN * LDBT];   // 18432 B, [n][k]
    int tid = threadIdx.x;
    int lane = tid & 31, wave = tid >> 5;
    int wm = wave & 3, wn = wave >> 2;          // 4x2 wave grid
    int lh = lane >> 4, ll = lane & 15;
    int m0 = blockIdx.y * BM;
    int n0 = blockIdx.x * BN;

    v8f acc[2][4];
    #pragma unroll
    for (int a = 0; a < 2; ++a)
        #pragma unroll
        for (int b = 0; b < 4; ++b)
            #pragma unroll
            for (int e = 0; e < 8; ++e) acc[a][b][e] = 0.f;

    for (int k0 = 0; k0 < K; k0 += BK) {
        // stage A tile (128x64) via CDNA5 async DMA: no VGPR round-trip
        for (int idx = tid; idx < (BM * BK) / 8; idx += 256) {   // 1024
            int r = idx >> 3, c8 = idx & 7;
            const unsigned short* gp = &A[(size_t)(m0 + r) * K + k0 + c8 * 8];
            async_copy_b128((unsigned int)(size_t)&As[r * LDA + c8 * 8], gp);
        }
        // stage B tile (64x128) transposed -> BsT[n][k] (needs in-flight transpose)
        for (int idx = tid; idx < (BK * BN) / 8; idx += 256) {   // 1024
            int r = idx >> 4, c8 = idx & 15;
            const unsigned short* gp = &B[(size_t)(k0 + r) * N + n0 + c8 * 8];
            Pack8 uu; uu.q = *(const v4u*)gp;
            if (k0 + BK < K) __builtin_prefetch(gp + (size_t)BK * N, 0, 2);
            #pragma unroll
            for (int j = 0; j < 8; ++j) BsT[(c8 * 8 + j) * LDBT + r] = uu.s[j];
        }
        async_wait0();
        __syncthreads();

        #pragma unroll
        for (int ks = 0; ks < 2; ++ks) {        // two 32-deep WMMA steps
            // A frags: e[0..7]=K(kb..), e[8..15]=K(kb+16..), kb=lh*8 (ISA bf16 A layout)
            FragBF af[2];
            #pragma unroll
            for (int mt = 0; mt < 2; ++mt) {
                const unsigned short* ap = &As[(wm * 32 + mt * 16 + ll) * LDA + ks * 32 + lh * 8];
                af[mt].q[0] = *(const v4u*)ap;
                af[mt].q[1] = *(const v4u*)(ap + 16);
            }
            // B frags: lane col = ll, e[0..7]=B[kb..][col], e[8..15]=B[kb+16..][col], kb=lh*8
            FragBF bf[4];
            #pragma unroll
            for (int nt = 0; nt < 4; ++nt) {
                const unsigned short* bp = &BsT[(wn * 64 + nt * 16 + ll) * LDBT + ks * 32 + lh * 8];
                bf[nt].q[0] = *(const v4u*)bp;
                bf[nt].q[1] = *(const v4u*)(bp + 16);
            }
            #pragma unroll
            for (int mt = 0; mt < 2; ++mt)
                #pragma unroll
                for (int nt = 0; nt < 4; ++nt)
                    acc[mt][nt] = __builtin_amdgcn_wmma_f32_16x16x32_bf16(
                        false, af[mt].v, false, bf[nt].v, (short)0, acc[mt][nt], false, false);
        }
        __syncthreads();
    }
    // store C (row = v + 8*lh, col = ll per 16-wide tile)
    #pragma unroll
    for (int mt = 0; mt < 2; ++mt)
        #pragma unroll
        for (int nt = 0; nt < 4; ++nt) {
            int col = n0 + wn * 64 + nt * 16 + ll;
            #pragma unroll
            for (int v = 0; v < 8; ++v) {
                int row = m0 + wm * 32 + mt * 16 + v + 8 * lh;
                C[(size_t)row * N + col] = acc[mt][nt][v];
            }
        }
}

// ---------------- 3a. Q RMSNorm + RoPE -> bf16 [H][T][HD] ----------------
__global__ void k_qnorm_rope(const float* __restrict__ Qf,   // (T, NH*HD)
                             const float* __restrict__ w,
                             const float* __restrict__ cosb,
                             const float* __restrict__ sinb,
                             unsigned short* __restrict__ Qb)
{
    __shared__ float ys[HD];
    __shared__ float red[4];
    int blk = blockIdx.x;          // T*NH
    int t = blk / NH, h = blk % NH;
    int i = threadIdx.x;           // 128
    float x = Qf[((size_t)t * NH + h) * HD + i];
    float ss = x * x;
    #pragma unroll
    for (int m = 16; m >= 1; m >>= 1) ss += __shfl_xor(ss, m, 32);
    if ((i & 31) == 0) red[i >> 5] = ss;
    __syncthreads();
    float inv = rsqrtf((red[0] + red[1] + red[2] + red[3]) / (float)HD + 1e-6f);
    float y = x * inv * w[i];
    ys[i] = y;
    __syncthreads();
    float rot = (i < 64) ? -ys[i + 64] : ys[i - 64];
    float o = y * cosb[t * HD + i] + rot * sinb[t * HD + i];
    Qb[((size_t)h * T_SEQ + t) * HD + i] = f2bf(o);
}

// ---------------- 3b. K RMSNorm+RoPE and V convert -> rows [RV, KV_LEN) ----------------
__global__ void k_knorm_rope_v(const float* __restrict__ Kf,  // (T, NKV*HD)
                               const float* __restrict__ Vf,
                               const float* __restrict__ w,
                               const float* __restrict__ cosb,
                               const float* __restrict__ sinb,
                               unsigned short* __restrict__ Kall,
                               unsigned short* __restrict__ Vall)
{
    __shared__ float ys[HD];
    __shared__ float red[4];
    int blk = blockIdx.x;           // T*NKV
    int t = blk / NKV, h = blk % NKV;
    int i = threadIdx.x;
    float x = Kf[((size_t)t * NKV + h) * HD + i];
    float ss = x * x;
    #pragma unroll
    for (int m = 16; m >= 1; m >>= 1) ss += __shfl_xor(ss, m, 32);
    if ((i & 31) == 0) red[i >> 5] = ss;
    __syncthreads();
    float inv = rsqrtf((red[0] + red[1] + red[2] + red[3]) / (float)HD + 1e-6f);
    float y = x * inv * w[i];
    ys[i] = y;
    __syncthreads();
    float rot = (i < 64) ? -ys[i + 64] : ys[i - 64];
    float o = y * cosb[t * HD + i] + rot * sinb[t * HD + i];
    size_t dsti = ((size_t)h * KV_LEN + RV + t) * HD + i;
    Kall[dsti] = f2bf(o);
    Vall[dsti] = f2bf(Vf[((size_t)t * NKV + h) * HD + i]);
}

// ---------------- 4. flash attention (WMMA, online softmax) ----------------
#define QT   64
#define CK   64
#define LDK  (HD + 8)   // 136 shorts, 272 B rows (16B aligned)
#define LDVT (CK + 8)   // 72 shorts, 144 B rows
#define LDP  (CK + 8)   // 72

__global__ __launch_bounds__(128) void k_attn(
    const unsigned short* __restrict__ Qb,    // (NH, T, HD)
    const unsigned short* __restrict__ Kall,  // (NKV, KV_LEN, HD)
    const unsigned short* __restrict__ Vall,
    unsigned short* __restrict__ AO)          // (T, NH*HD)
{
    __shared__ unsigned short Ks[CK * LDK];        // 17408 B  [key][d]
    __shared__ unsigned short VsT[HD * LDVT];      // 18432 B  [d][key]
    __shared__ unsigned short Ps[4 * 16 * LDP];    //  9216 B

    int h  = blockIdx.x / (T_SEQ / QT);
    int qt = blockIdx.x % (T_SEQ / QT);
    int qbase = qt * QT;
    int kvh = h / GQA;
    int tid = threadIdx.x;
    int lane = tid & 31, wave = tid >> 5;
    int lh = lane >> 4, ll = lane & 15;
    const float scaling = 0.08838834764831845f;   // 128^-0.5

    // per-wave Q row block: A-fragments for 4 k-steps of 32 (b128 loads)
    FragBF qf[4];
    {
        int trow = qbase + wave * 16 + ll;
        const unsigned short* qp = Qb + ((size_t)h * T_SEQ + trow) * HD;
        #pragma unroll
        for (int ks = 0; ks < 4; ++ks) {
            const unsigned short* p = qp + ks * 32 + lh * 8;
            qf[ks].q[0] = *(const v4u*)p;
            qf[ks].q[1] = *(const v4u*)(p + 16);
        }
    }

    v8f o[8];
    #pragma unroll
    for (int d = 0; d < 8; ++d)
        #pragma unroll
        for (int e = 0; e < 8; ++e) o[d][e] = 0.f;
    float mrow[8], lrow[8];
    #pragma unroll
    for (int v = 0; v < 8; ++v) { mrow[v] = -1e30f; lrow[v] = 0.f; }

    int nch = qt + 2;   // keys needed: [0, RV + qbase + QT); RV == CK
    for (int c = 0; c < nch; ++c) {
        int kb0 = c * CK;
        // stage K row-major via async DMA; V transposed via VGPR scatter
        for (int idx = tid; idx < (CK * HD) / 8; idx += 128) {   // 1024
            int r = idx >> 4, c8 = idx & 15;
            size_t g = ((size_t)kvh * KV_LEN + kb0 + r) * HD + c8 * 8;
            async_copy_b128((unsigned int)(size_t)&Ks[r * LDK + c8 * 8], &Kall[g]);
            Pack8 uu; uu.q = *(const v4u*)&Vall[g];
            #pragma unroll
            for (int j = 0; j < 8; ++j) VsT[(c8 * 8 + j) * LDVT + r] = uu.s[j];
        }
        async_wait0();
        __syncthreads();

        // S = Q * K^T : 4 n-tiles of 16 keys, b128 fragment loads
        v8f sacc[4];
        #pragma unroll
        for (int nt = 0; nt < 4; ++nt) {
            #pragma unroll
            for (int e = 0; e < 8; ++e) sacc[nt][e] = 0.f;
            #pragma unroll
            for (int ks = 0; ks < 4; ++ks) {
                FragBF b;
                const unsigned short* kp = &Ks[(nt * 16 + ll) * LDK + ks * 32 + lh * 8];
                b.q[0] = *(const v4u*)kp;
                b.q[1] = *(const v4u*)(kp + 16);
                sacc[nt] = __builtin_amdgcn_wmma_f32_16x16x32_bf16(
                    false, qf[ks].v, false, b.v, (short)0, sacc[nt], false, false);
            }
        }

        // scale + causal mask + online softmax (rows rm = v + 8*lh, col = ll per n-tile)
        float pv[4][8], cmax[8];
        #pragma unroll
        for (int v = 0; v < 8; ++v) cmax[v] = -1e30f;
        #pragma unroll
        for (int nt = 0; nt < 4; ++nt) {
            int kg = kb0 + nt * 16 + ll;
            #pragma unroll
            for (int v = 0; v < 8; ++v) {
                int qg = qbase + wave * 16 + v + 8 * lh;
                float sv = sacc[nt][v] * scaling;
                if (kg >= RV && (kg - RV) > qg) sv = -1e30f;
                pv[nt][v] = sv;
                cmax[v] = fmaxf(cmax[v], sv);
            }
        }
        float alpha[8];
        #pragma unroll
        for (int v = 0; v < 8; ++v) {
            float x = cmax[v];
            x = fmaxf(x, __shfl_xor(x, 1, 32));
            x = fmaxf(x, __shfl_xor(x, 2, 32));
            x = fmaxf(x, __shfl_xor(x, 4, 32));
            x = fmaxf(x, __shfl_xor(x, 8, 32));
            float nm = fmaxf(mrow[v], x);
            alpha[v] = __expf(mrow[v] - nm);
            mrow[v] = nm;
            float s = 0.f;
            #pragma unroll
            for (int nt = 0; nt < 4; ++nt) {
                float p = __expf(pv[nt][v] - nm);
                pv[nt][v] = p;
                s += p;
            }
            s += __shfl_xor(s, 1, 32); s += __shfl_xor(s, 2, 32);
            s += __shfl_xor(s, 4, 32); s += __shfl_xor(s, 8, 32);
            lrow[v] = lrow[v] * alpha[v] + s;
        }
        // rescale O accumulators
        #pragma unroll
        for (int d = 0; d < 8; ++d)
            #pragma unroll
            for (int v = 0; v < 8; ++v) o[d][v] *= alpha[v];

        // P (C-layout) -> LDS -> A-layout
        unsigned short* pw = &Ps[wave * 16 * LDP];
        #pragma unroll
        for (int nt = 0; nt < 4; ++nt)
            #pragma unroll
            for (int v = 0; v < 8; ++v)
                pw[(v + 8 * lh) * LDP + nt * 16 + ll] = f2bf(pv[nt][v]);
        __syncthreads();

        // O += P * V : K-dim 64 = 2 wmma steps, 8 d-tiles
        #pragma unroll
        for (int ks2 = 0; ks2 < 2; ++ks2) {
            FragBF a;
            const unsigned short* pr = &Ps[(wave * 16 + ll) * LDP + ks2 * 32 + lh * 8];
            a.q[0] = *(const v4u*)pr;
            a.q[1] = *(const v4u*)(pr + 16);
            #pragma unroll
            for (int d = 0; d < 8; ++d) {
                FragBF b;
                const unsigned short* vp = &VsT[(d * 16 + ll) * LDVT + ks2 * 32 + lh * 8];
                b.q[0] = *(const v4u*)vp;
                b.q[1] = *(const v4u*)(vp + 16);
                o[d] = __builtin_amdgcn_wmma_f32_16x16x32_bf16(
                    false, a.v, false, b.v, (short)0, o[d], false, false);
            }
        }
        __syncthreads();
    }

    // epilogue: O / l  -> AO[t][h*HD + d]
    #pragma unroll
    for (int d = 0; d < 8; ++d)
        #pragma unroll
        for (int v = 0; v < 8; ++v) {
            int row = qbase + wave * 16 + v + 8 * lh;
            int col = h * HD + d * 16 + ll;
            AO[(size_t)row * (NH * HD) + col] = f2bf(o[d][v] / lrow[v]);
        }
}

// ---------------- launch ----------------
extern "C" void kernel_launch(void* const* d_in, const int* in_sizes, int n_in,
                              void* d_out, int out_size, void* d_ws, size_t ws_size,
                              hipStream_t stream)
{
    const float* hs      = (const float*)d_in[0];
    const float* vkeys   = (const float*)d_in[1];
    const float* vvals   = (const float*)d_in[2];
    const float* Wq      = (const float*)d_in[3];
    const float* Wk      = (const float*)d_in[4];
    const float* Wv      = (const float*)d_in[5];
    const float* Wo      = (const float*)d_in[6];
    const float* q_norm  = (const float*)d_in[7];
    const float* k_norm  = (const float*)d_in[8];
    const float* lkA     = (const float*)d_in[9];
    const float* lkB     = (const float*)d_in[10];
    const float* lvA     = (const float*)d_in[11];
    const float* lvB     = (const float*)d_in[12];
    const float* scale_k = (const float*)d_in[13];
    const float* scale_v = (const float*)d_in[14];
    // d_in[15] = attention_mask (causal; computed analytically in-kernel)
    const float* cosb    = (const float*)d_in[16];
    const float* sinb    = (const float*)d_in[17];
    float* out = (float*)d_out;

    unsigned char* base = (unsigned char*)d_ws;
    size_t off = 0;
    auto alloc = [&](size_t bytes) -> void* {
        void* p = base + off;
        off = (off + bytes + 255) & ~(size_t)255;
        return p;
    };
    const size_t N_TD = (size_t)T_SEQ * D_MODEL;          // 8388608
    const size_t N_QQ = (size_t)D_MODEL * D_MODEL;        // 16777216
    const size_t N_KV = (size_t)D_MODEL * NKV * HD;       // 4194304
    unsigned short* Xb   = (unsigned short*)alloc(N_TD * 2);
    unsigned short* Wqb  = (unsigned short*)alloc(N_QQ * 2);
    unsigned short* Wkb  = (unsigned short*)alloc(N_KV * 2);
    unsigned short* Wvb  = (unsigned short*)alloc(N_KV * 2);
    unsigned short* Wob  = (unsigned short*)alloc(N_QQ * 2);
    float*          Qf   = (float*)alloc(N_TD * 4);
    float*          Kf   = (float*)alloc((size_t)T_SEQ * NKV * HD * 4);
    float*          Vf   = (float*)alloc((size_t)T_SEQ * NKV * HD * 4);
    unsigned short* Qb   = (unsigned short*)alloc((size_t)NH * T_SEQ * HD * 2);
    unsigned short* Kall = (unsigned short*)alloc((size_t)NKV * KV_LEN * HD * 2);
    unsigned short* Vall = (unsigned short*)alloc((size_t)NKV * KV_LEN * HD * 2);
    unsigned short* AO   = (unsigned short*)alloc(N_TD * 2);
    (void)ws_size; (void)in_sizes; (void)n_in; (void)out_size;

    // 0) converts
    k_cvt_bf16<<<2048, 256, 0, stream>>>(hs, Xb, (int)N_TD);
    k_cvt_bf16<<<2048, 256, 0, stream>>>(Wq, Wqb, (int)N_QQ);
    k_cvt_bf16<<<2048, 256, 0, stream>>>(Wk, Wkb, (int)N_KV);
    k_cvt_bf16<<<2048, 256, 0, stream>>>(Wv, Wvb, (int)N_KV);
    k_cvt_bf16<<<2048, 256, 0, stream>>>(Wo, Wob, (int)N_QQ);

    // 1) LoRA adapter -> virtual rows of Kall/Vall
    k_lora_adapt<<<NKV * RV, HD, 0, stream>>>(vkeys, lkA, lkB, scale_k, Kall);
    k_lora_adapt<<<NKV * RV, HD, 0, stream>>>(vvals, lvA, lvB, scale_v, Vall);

    // 2) projections (WMMA GEMM)
    {
        dim3 gq(D_MODEL / BN, T_SEQ / BM);                 // (32,16)
        k_gemm_bf16<<<gq, 256, 0, stream>>>(Xb, Wqb, Qf, T_SEQ, D_MODEL, D_MODEL);
        dim3 gk((NKV * HD) / BN, T_SEQ / BM);              // (8,16)
        k_gemm_bf16<<<gk, 256, 0, stream>>>(Xb, Wkb, Kf, T_SEQ, NKV * HD, D_MODEL);
        k_gemm_bf16<<<gk, 256, 0, stream>>>(Xb, Wvb, Vf, T_SEQ, NKV * HD, D_MODEL);
    }

    // 3) norms + rope + pack
    k_qnorm_rope<<<T_SEQ * NH, HD, 0, stream>>>(Qf, q_norm, cosb, sinb, Qb);
    k_knorm_rope_v<<<T_SEQ * NKV, HD, 0, stream>>>(Kf, Vf, k_norm, cosb, sinb, Kall, Vall);

    // 4) attention
    k_attn<<<NH * (T_SEQ / QT), 128, 0, stream>>>(Qb, Kall, Vall, AO);

    // 5) output projection
    {
        dim3 go(D_MODEL / BN, T_SEQ / BM);
        k_gemm_bf16<<<go, 256, 0, stream>>>(AO, Wob, out, T_SEQ, D_MODEL, D_MODEL);
    }
}
